// Conv1DLocal_11416023073152
// MI455X (gfx1250) — matrix operand order repaired and checked by reading, not compiled, not run on hardware
//
#include <hip/hip_runtime.h>

// CDNA5 / gfx1250 depthwise conv1d (C=128, K=128, SAME) + softplus via WMMA.
//
// Per channel c the conv is a Toeplitz matvec; we lift it to dense 16x16
// matmuls with a *banded* weight matrix  B[s,n] = w[s-n, c]  (zero outside
// the band) and  A[m,s] = x_c[t0' + 16m + s], s in [0,160).  Then
//   D[m,n] = sum_j x_c[t0' + 16m + n + j] * w[j]
// i.e. the pre-activation at output time t0' + 63 + 16m + n (pad_left = 63).
// K=160 -> 5 chained v_wmma_f32_16x16x32_f16  (80% useful-MAC efficiency).

typedef __attribute__((ext_vector_type(16))) _Float16 v16h;
typedef __attribute__((ext_vector_type(8)))  float    v8f;

#define T_LEN     32768
#define NCH       128
#define KSZ       128
#define NBATCH    16

#define CH_PER_WG 16      // channels per workgroup == waves per workgroup
#define THREADS   512     // 16 wave32
#define TB_PER_WG 8       // time blocks per workgroup (amortize weight setup)
#define TBLOCK    256     // output times per block (16x16 D tile chain)
#define XWIN      400     // input window: 240 (row stride) + 160 (band width)
#define PAD_LEFT  63      // SAME, K=128 -> pad_left = 63, pad_right = 64

#define XPITCH    408     // halves per channel row in xs (mult of 8 -> 16B align)
#define WPITCH    180     // halves per channel row in wlds (16 zero + 128 w + pad)
#define OPITCH    17      // floats per t-row in outstage (bank stagger)

union AV { uint4 u[2]; v16h v; };
union BV { v16h  v; _Float16 h[16]; };

// Raw hardware transcendentals: the libm versions drag in denorm-guard
// cndmask/ldexp sequences that are provably dead here (log arg >= 1.0).
// v_nop after each TRANS op satisfies the CDNA5 TRANS-result hazard (ISA 7.4).
__device__ __forceinline__ float raw_exp2(float a) {
  float r;
  asm volatile("v_exp_f32 %0, %1\n\tv_nop" : "=v"(r) : "v"(a));
  return r;
}
__device__ __forceinline__ float raw_log2(float a) {
  float r;
  asm volatile("v_log_f32 %0, %1\n\tv_nop" : "=v"(r) : "v"(a));
  return r;
}

__device__ __forceinline__ float fast_softplus(float v) {
  // softplus(v) = max(v,0) + log(1 + exp(-|v|))
  const float NLOG2E = -1.4426950408889634f;  // -log2(e)
  const float LN2    =  0.6931471805599453f;
  float e = raw_exp2(fabsf(v) * NLOG2E);      // exp(-|v|) in (0, 1]
  float l = raw_log2(1.0f + e);               // arg in (1, 2] -> never denormal
  return fmaxf(v, 0.0f) + l * LN2;
}

__global__ __launch_bounds__(THREADS)
void dwconv_softplus_wmma(const float* __restrict__ x,
                          const float* __restrict__ w,
                          float* __restrict__ out) {
  __shared__ __align__(16) _Float16 xs[CH_PER_WG * XPITCH];      // x window, f16
  __shared__ _Float16 wlds[CH_PER_WG * WPITCH];                  // guarded weights
  __shared__ float    outstage[TBLOCK * OPITCH];                 // store transpose

  const int tid = threadIdx.x;
  const int b   = blockIdx.z;                 // batch
  const int cg  = blockIdx.y;                 // channel group (16 channels)
  const int tbg = blockIdx.x;                 // group of 8 time blocks
  const int c0  = cg * CH_PER_WG;

  // ---- stage weights once: zero guard margins, then fill w[j, c] ----
  for (int i = tid; i < CH_PER_WG * WPITCH; i += THREADS)
    wlds[i] = (_Float16)0.0f;
  __syncthreads();
  for (int i = tid; i < CH_PER_WG * KSZ; i += THREADS) {
    int cc = i & (CH_PER_WG - 1);
    int j  = i >> 4;
    wlds[cc * WPITCH + 16 + j] = (_Float16)w[j * NCH + c0 + cc];
  }
  __syncthreads();

  // ---- per-lane WMMA operand geometry (wave32) ----
  const int lane = tid & 31;
  const int wave = tid >> 5;                  // channel within group
  const int n    = lane & 15;                 // column (B/N) or row (A/M) index
  const int hi   = lane >> 4;                 // lane half
  const int a_k0 = hi ? 8 : 0;                // A: K in {k0..k0+7, k0+16..k0+23}
  const int b_k0 = hi ? 8 : 0;                // B: K in {b_k0..b_k0+7, b_k0+16..b_k0+23}

  // ---- build banded-weight B tiles in registers (5 tiles of K=32) ----
  v16h Bt[5];
  {
    const _Float16* wb = &wlds[wave * WPITCH];
    #pragma unroll
    for (int q = 0; q < 5; ++q) {
      BV t;
      const int base = 16 + 32 * q + b_k0 - n;   // in [1, 176) -> always in bounds
      #pragma unroll
      for (int e = 0; e < 16; ++e) t.h[e] = wb[base + e + (e & 8)];   // B[k,n] = w[s-n]
      Bt[q] = t.v;
    }
  }

  for (int i = 0; i < TB_PER_WG; ++i) {
    const int tb    = tbg * TB_PER_WG + i;
    const int tout0 = tb * TBLOCK;
    const int t0p   = tout0 - PAD_LEFT;

    __syncthreads();   // protect xs (prev WMMA reads) and outstage (prev stores)

    // ---- stage x window: 16 ch x 400 t, float4 (b128) loads, f16 in LDS ----
    // 4 consecutive channels share one float4; guard is per-t only.
    for (int idx = tid; idx < XWIN * 4; idx += THREADS) {
      int cc4 = idx & 3;                      // which float4 within 16 channels
      int tl  = idx >> 2;                     // local t in [0, 400)
      int t   = t0p + tl;
      float4 v = make_float4(0.f, 0.f, 0.f, 0.f);
      if (t >= 0 && t < T_LEN)
        v = *reinterpret_cast<const float4*>(
              &x[((size_t)b * T_LEN + t) * NCH + c0 + 4 * cc4]);
      _Float16* d = &xs[(4 * cc4) * XPITCH + tl];
      d[0 * XPITCH] = (_Float16)v.x;
      d[1 * XPITCH] = (_Float16)v.y;
      d[2 * XPITCH] = (_Float16)v.z;
      d[3 * XPITCH] = (_Float16)v.w;
    }

    // ---- prefetch next block's window while this block computes ----
    if (i + 1 < TB_PER_WG && tid < XWIN) {
      int t = t0p + TBLOCK + tid;
      if (t >= 0 && t < T_LEN)
        __builtin_prefetch(&x[((size_t)b * T_LEN + t) * NCH + c0], 0, 1);
    }
    __syncthreads();

    // ---- 5 chained WMMAs: D = A(16x160) * B(160x16), K=32 each ----
    v8f acc = {0.f, 0.f, 0.f, 0.f, 0.f, 0.f, 0.f, 0.f};
    const _Float16* xrow = &xs[wave * XPITCH];
    #pragma unroll
    for (int q = 0; q < 5; ++q) {
      AV a;
      // halves offset 16*m + 32*q + a_k0 is a multiple of 8 -> 16B aligned
      const uint4* p = reinterpret_cast<const uint4*>(xrow + 16 * n + 32 * q + a_k0);
      a.u[0] = p[0];          // K = a_k0 .. a_k0+7
      a.u[1] = p[2];          // K = a_k0+16 .. a_k0+23
      acc = __builtin_amdgcn_wmma_f32_16x16x32_f16(
          /*neg_a=*/false, a.v, /*neg_b=*/false, Bt[q],
          /*c_mod=*/(short)0, acc, /*reuse_a=*/false, /*reuse_b=*/false);
    }

    // ---- softplus + LDS transpose: D[m,n] -> outstage[16*m + n][wave] ----
    #pragma unroll
    for (int r = 0; r < 8; ++r) {
      int tl = 16 * (r + 8 * hi) + n;          // local output time
      outstage[tl * OPITCH + wave] = fast_softplus(acc[r]);
    }
    __syncthreads();

    // ---- coalesced float4 store: [256 t][16 c] slab of out[b, t, c] ----
    for (int idx = tid; idx < TBLOCK * 4; idx += THREADS) {   // 2 iterations
      int cc4 = idx & 3;
      int tl  = idx >> 2;
      const float* s = &outstage[tl * OPITCH + 4 * cc4];
      float4 v = make_float4(s[0], s[1], s[2], s[3]);
      *reinterpret_cast<float4*>(
          &out[((size_t)b * T_LEN + tout0 + tl) * NCH + c0 + 4 * cc4]) = v;
    }
  }
}

extern "C" void kernel_launch(void* const* d_in, const int* in_sizes, int n_in,
                              void* d_out, int out_size, void* d_ws, size_t ws_size,
                              hipStream_t stream) {
  const float* x = (const float*)d_in[0];   // [16, 32768, 128] f32
  const float* w = (const float*)d_in[1];   // [128, 1, 128]    f32
  float* out     = (float*)d_out;           // [16, 32768, 128] f32

  dim3 grid((T_LEN / TBLOCK) / TB_PER_WG,   // 16 time-block groups
            NCH / CH_PER_WG,                // 8 channel groups
            NBATCH);                        // 16 batches
  dwconv_softplus_wmma<<<grid, THREADS, 0, stream>>>(x, w, out);
}
